// ChannelSelfAttentionModule_10307921511018
// MI455X (gfx1250) — matrix operand order repaired and checked by reading, not compiled, not run on hardware
//
#include <hip/hip_runtime.h>
#include <hip/hip_bf16.h>
#include <math.h>

// ---------------------------------------------------------------------------
// CSAM forward for MI455X (gfx1250, wave32, WMMA).
// B=4, C=64, H=W=64 -> N=4096 pixels/batch, ROWS = B*N = 16384.
// Attention (2 * 2*B*N*N*C ~ 17 GFLOP) dominates -> fp16 WMMA flash attention.
// This revision computes S^T = K.Q^T so the softmax axis lives in registers
// (one shfl_xor(16) per reduction instead of 4 ds_bpermute chains per row),
// folds 1/sqrt(C) into the q projection, and gives each wave 32 n-rows so
// K/V fragments are reused twice (2x arithmetic intensity vs L2).
// ---------------------------------------------------------------------------

#define B_   4
#define C_   64
#define N_   4096
#define ROWS (B_ * N_)   // 16384

typedef __attribute__((ext_vector_type(16))) _Float16 v16h;
typedef __attribute__((ext_vector_type(8)))  float    v8f;

union FragU {
  v16h      h;
  unsigned  u[8];
  _Float16  e[16];
};

union Pack2 {
  unsigned  u;
  _Float16  e[2];
};

__device__ inline v8f v8f_zero() {
  v8f z;
#pragma unroll
  for (int i = 0; i < 8; ++i) z[i] = 0.0f;
  return z;
}

__device__ inline v8f wmma16(v16h a, v16h b, v8f c) {
  // D = A(16x32 f16) * B(32x16 f16) + C(16x16 f32)
  return __builtin_amdgcn_wmma_f32_16x16x32_f16(
      /*neg_a=*/false, a, /*neg_b=*/false, b,
      /*c_mod=*/(short)0, c, /*reuse_a=*/false, /*reuse_b=*/false);
}

// A fragment (16 rows x 32 K, f16) from row-major [row][K] f16 storage.
// ISA 7.12.2: lane L -> M = L%16; VGPR v holds K pair:
//   kb = k0 + (v>=4 ? 16:0) + (L>=16 ? 8:0) + (v&3)*2   (K even/odd in one dword)
__device__ inline v16h load_frag_a(const unsigned* base_u, int row0,
                                   int row_stride_u, int k0, int lane) {
  FragU f;
  const int rl = lane & 15;
  const int hi = lane >> 4;
  const unsigned* rp = base_u + (long)(row0 + rl) * row_stride_u;
#pragma unroll
  for (int v = 0; v < 8; ++v) {
    int kb = k0 + ((v & 4) ? 16 : 0) + (hi ? 8 : 0) + (v & 3) * 2;
    f.u[v] = rp[kb >> 1];
  }
  return f.h;
}

// B fragment (32 K x 16 cols, f16) from [col][K] f16 storage.
// ISA 7.12.2: lane L -> N = L%16; VGPR v holds K rows k0 + (v>=4?16:0) + (L<16?0:8) + (v&3)*2, +1
__device__ inline v16h load_frag_b(const unsigned* base_u, int col0,
                                   long col_stride_u, int k0, int lane) {
  FragU f;
  const int cl = lane & 15;
  const int hk = (lane >> 4) ? 8 : 0;
  const unsigned* cp = base_u + (long)(col0 + cl) * col_stride_u;
#pragma unroll
  for (int v = 0; v < 8; ++v) {
    int k = k0 + ((v & 4) ? 16 : 0) + hk + (v & 3) * 2;
    f.u[v] = cp[k >> 1];
  }
  return f.h;
}

// ---------------------------------------------------------------------------
// Weight f32 -> f16 convert (with optional scale, used to fold 1/sqrt(C))
// ---------------------------------------------------------------------------
__global__ __launch_bounds__(256)
void f32_to_f16_kernel(const float* __restrict__ in, _Float16* __restrict__ out,
                       int n, float scale) {
  int t = blockIdx.x * 256 + threadIdx.x;
  if (t < n) out[t] = (_Float16)(in[t] * scale);
}

// ---------------------------------------------------------------------------
// LayerNorm over C for x in [B][C][N] layout.
// Writes xn (f32, same layout, feeds dwconvs) and xn_h (f16 [row][C]).
// ---------------------------------------------------------------------------
__global__ __launch_bounds__(256)
void ln1_kernel(const float* __restrict__ x, const float* __restrict__ g,
                const float* __restrict__ be, float* __restrict__ xn,
                _Float16* __restrict__ xnh) {
  int t = blockIdx.x * 256 + threadIdx.x;      // over B*N, coalesced in p
  int b = t >> 12;
  int p = t & (N_ - 1);
  const float* xb = x + ((long)b * C_) * N_ + p;
  float xv[C_];
  float s = 0.f;
#pragma unroll
  for (int c = 0; c < C_; ++c) { float v = xb[(long)c * N_]; xv[c] = v; s += v; }
  float mu = s * (1.0f / C_);
  float vs = 0.f;
#pragma unroll
  for (int c = 0; c < C_; ++c) { float d = xv[c] - mu; vs += d * d; }
  float inv = rsqrtf(vs * (1.0f / C_) + 1e-5f);
  float* xnb = xn + ((long)b * C_) * N_ + p;
  _Float16* xh = xnh + (long)t * C_;
#pragma unroll
  for (int c = 0; c < C_; ++c) {
    float v = (xv[c] - mu) * inv * g[c] + be[c];
    xnb[(long)c * N_] = v;
    xh[c] = (_Float16)v;
  }
}

// LayerNorm over C for x_att stored [row][C] f32 -> xn2_h f16 [row][C].
__global__ __launch_bounds__(256)
void ln2_kernel(const float* __restrict__ xatt, const float* __restrict__ g,
                const float* __restrict__ be, _Float16* __restrict__ xnh) {
  int t = blockIdx.x * 256 + threadIdx.x;      // over B*N
  const float* r = xatt + (long)t * C_;
  float s = 0.f;
#pragma unroll
  for (int c = 0; c < C_; ++c) s += r[c];
  float mu = s * (1.0f / C_);
  float vs = 0.f;
#pragma unroll
  for (int c = 0; c < C_; ++c) { float d = r[c] - mu; vs += d * d; }
  float inv = rsqrtf(vs * (1.0f / C_) + 1e-5f);
  _Float16* xh = xnh + (long)t * C_;
#pragma unroll
  for (int c = 0; c < C_; ++c)
    xh[c] = (_Float16)((r[c] - mu) * inv * g[c] + be[c]);
}

// ---------------------------------------------------------------------------
// Depthwise 3x3 (SAME) on xn f32 [B][C][N]; output f16, either
// transposed [b][p][c] (A/B [row][K] layout for attention) or [b][c][p].
// ---------------------------------------------------------------------------
__global__ __launch_bounds__(256)
void dwconv64_kernel(const float* __restrict__ xin, const float* __restrict__ w,
                     const float* __restrict__ bias, _Float16* __restrict__ out,
                     int transpose_out) {
  int t = blockIdx.x * 256 + threadIdx.x;      // B*C*N threads
  int p = t & (N_ - 1);
  int c = (t >> 12) & (C_ - 1);
  int b = t >> 18;
  int h = p >> 6, wx = p & 63;
  const float* xc = xin + ((long)b * C_ + c) * N_;
  float s = bias[c];
#pragma unroll
  for (int ky = 0; ky < 3; ++ky) {
    int ih = h + ky - 1;
    if (ih < 0 || ih >= 64) continue;
#pragma unroll
    for (int kx = 0; kx < 3; ++kx) {
      int iw = wx + kx - 1;
      if (iw < 0 || iw >= 64) continue;
      s += xc[ih * 64 + iw] * w[c * 9 + ky * 3 + kx];
    }
  }
  if (transpose_out) out[((long)b * N_ + p) * C_ + c] = (_Float16)s;
  else               out[((long)b * C_ + c) * N_ + p] = (_Float16)s;
}

// ---------------------------------------------------------------------------
// Depthwise 3x3 (SAME) on f16 [row][128] + exact GELU; optional branch multiply.
// ---------------------------------------------------------------------------
__global__ __launch_bounds__(256)
void dwconv128_gelu_kernel(const _Float16* __restrict__ hin, const float* __restrict__ w,
                           const float* __restrict__ bias, const _Float16* __restrict__ mult,
                           _Float16* __restrict__ out) {
  long t = (long)blockIdx.x * 256 + threadIdx.x;   // B*N*128 threads
  int c = (int)(t & 127);
  long rowg = t >> 7;
  int p = (int)(rowg & (N_ - 1));
  long b = rowg >> 12;
  int h = p >> 6, wx = p & 63;
  float s = bias[c];
#pragma unroll
  for (int ky = 0; ky < 3; ++ky) {
    int ih = h + ky - 1;
    if (ih < 0 || ih >= 64) continue;
#pragma unroll
    for (int kx = 0; kx < 3; ++kx) {
      int iw = wx + kx - 1;
      if (iw < 0 || iw >= 64) continue;
      s += (float)hin[(b * N_ + (long)(ih * 64 + iw)) * 128 + c] * w[c * 9 + ky * 3 + kx];
    }
  }
  float gl = 0.5f * s * (1.0f + erff(s * 0.70710678118654752f));   // exact GELU
  if (mult) gl *= (float)mult[t];
  out[t] = (_Float16)gl;
}

// ---------------------------------------------------------------------------
// Generic WMMA GEMM (1x1 conv): out[row][co] = act[row][:CIN].w[co][:CIN] + bias
// OM: 0 = f16 out [row][COUT]; 1 = f32 out [row][COUT]; 2 = f32 out [b][co][n]
// RM: 0 = none; 1 = + res f32 [b][co][n]; 2 = + res f32 [row][co]
// ---------------------------------------------------------------------------
template <int CIN, int COUT, int OM, int RM>
__global__ __launch_bounds__(128)
void gemm_wmma_kernel(const _Float16* __restrict__ act, const _Float16* __restrict__ w,
                      const float* __restrict__ bias, void* __restrict__ outp,
                      const float* __restrict__ res, float bias_scale) {
  const int lane = threadIdx.x & 31;
  const int wv   = threadIdx.x >> 5;
  const int row0 = (blockIdx.x * 4 + wv) * 16;
  const unsigned* act_u = (const unsigned*)act;
  const unsigned* w_u   = (const unsigned*)w;

  v8f acc[COUT / 16];
#pragma unroll
  for (int t = 0; t < COUT / 16; ++t) acc[t] = v8f_zero();

#pragma unroll
  for (int kh = 0; kh < CIN / 32; ++kh) {
    v16h a = load_frag_a(act_u, row0, CIN / 2, kh * 32, lane);
#pragma unroll
    for (int t = 0; t < COUT / 16; ++t) {
      v16h b = load_frag_b(w_u, t * 16, CIN / 2, kh * 32, lane);
      acc[t] = wmma16(a, b, acc[t]);
    }
  }

  const int rl   = lane & 15;
  const int mofs = (lane >> 4) ? 8 : 0;
#pragma unroll
  for (int t = 0; t < COUT / 16; ++t) {
    int co = t * 16 + rl;
    float bco = bias[co] * bias_scale;
#pragma unroll
    for (int r = 0; r < 8; ++r) {
      int grow = row0 + r + mofs;            // global row (b*N + n)
      int b_ = grow >> 12;
      int nl = grow & (N_ - 1);
      float v = acc[t][r] + bco;
      if (RM == 1) v += res[((long)b_ * COUT + co) * N_ + nl];
      if (RM == 2) v += res[(long)grow * COUT + co];
      if (OM == 0)      ((_Float16*)outp)[(long)grow * COUT + co] = (_Float16)v;
      else if (OM == 1) ((float*)outp)[(long)grow * COUT + co] = v;
      else              ((float*)outp)[((long)b_ * COUT + co) * N_ + nl] = v;
    }
  }
}

// ---------------------------------------------------------------------------
// Flash attention, transposed-score formulation.
//   S^T[m][n] = sum_c k[m][c] * q'[n][c]        (q' already scaled by 1/8)
//   O[n][c]   = sum_m softmax_m(S)[n][m] * v[c][m]
// A = K tile (rows m) from k_h [b][m][c]; B = Q (cols n) from q_h [row][c];
// C/D of S^T: m in registers, n in lanes -> softmax reduction is per-lane
// over 16 regs + one shfl_xor(16). P^T -> A transpose via per-wave LDS
// (pair-packed u32). Per-row rescale/normalize factors broadcast via LDS.
// Each wave owns 32 n-rows: K/V fragments reused twice.
// ---------------------------------------------------------------------------
__global__ __launch_bounds__(128)
void flash_attn_kernel(const _Float16* __restrict__ qh, const _Float16* __restrict__ kh,
                       const _Float16* __restrict__ vh, _Float16* __restrict__ oh) {
  __shared__ unsigned ldsPu[4][2][16][17];   // per-wave P^T staging (u32 pairs)
  __shared__ float    ldsC [4][2][16];       // per-wave row-factor broadcast
  const int lane = threadIdx.x & 31;
  const int wv   = threadIdx.x >> 5;
  const int row0 = (blockIdx.x * 4 + wv) * 32;     // 32 n-rows per wave
  const int b    = row0 >> 12;
  const int rl   = lane & 15;
  const int hi   = lane >> 4;

  const unsigned* q_u = (const unsigned*)qh;
  const unsigned* k_u = (const unsigned*)kh + (long)b * N_ * (C_ / 2);
  const unsigned* v_u = (const unsigned*)vh + (long)b * C_ * (N_ / 2);

  // Q as B operand (cols = n, K = c), loop-invariant.
  v16h bQ[2][2];
#pragma unroll
  for (int nt = 0; nt < 2; ++nt)
#pragma unroll
    for (int kh2 = 0; kh2 < 2; ++kh2)
      bQ[nt][kh2] = load_frag_b(q_u, row0 + nt * 16, C_ / 2, kh2 * 32, lane);

  v8f acc[2][4];
#pragma unroll
  for (int nt = 0; nt < 2; ++nt)
#pragma unroll
    for (int t = 0; t < 4; ++t) acc[nt][t] = v8f_zero();
  float mrun[2] = { -1e30f, -1e30f };
  float lrun[2] = { 0.f, 0.f };

  for (int mt = 0; mt < N_; mt += 32) {
    if (mt + 32 < N_) {
      __builtin_prefetch(k_u + (long)(mt + 32 + lane) * (C_ / 2), 0, 1);
    }
    // K A-fragments for this m-chunk (shared by both n-tiles).
    v16h aK[2][2];
#pragma unroll
    for (int mi = 0; mi < 2; ++mi)
#pragma unroll
      for (int kh2 = 0; kh2 < 2; ++kh2)
        aK[mi][kh2] = load_frag_a(k_u, mt + mi * 16, C_ / 2, kh2 * 32, lane);

#pragma unroll
    for (int nt = 0; nt < 2; ++nt) {
      v8f st0 = v8f_zero(), st1 = v8f_zero();
      st0 = wmma16(aK[0][0], bQ[nt][0], st0);
      st0 = wmma16(aK[0][1], bQ[nt][1], st0);
      st1 = wmma16(aK[1][0], bQ[nt][0], st1);
      st1 = wmma16(aK[1][1], bQ[nt][1], st1);

      // Per-lane max over the 16 m-values of column n, then combine halves.
      float tmax = st0[0];
#pragma unroll
      for (int r = 1; r < 8; ++r) tmax = fmaxf(tmax, st0[r]);
#pragma unroll
      for (int r = 0; r < 8; ++r) tmax = fmaxf(tmax, st1[r]);
      tmax = fmaxf(tmax, __shfl_xor(tmax, 16, 32));
      float mnew = fmaxf(mrun[nt], tmax);
      float corr = __expf(mrun[nt] - mnew);
      mrun[nt] = mnew;

      float p0[8], p1[8], ls = 0.f;
#pragma unroll
      for (int r = 0; r < 8; ++r) {
        p0[r] = __expf(st0[r] - mnew);
        p1[r] = __expf(st1[r] - mnew);
        ls += p0[r] + p1[r];
      }
      ls += __shfl_xor(ls, 16, 32);
      lrun[nt] = lrun[nt] * corr + ls;

      // Store P^T: lane holds column n = rl, rows m = r + 8*hi (+16 for st1).
      // Consecutive r are consecutive m -> pack pairs into u32.
#pragma unroll
      for (int r2 = 0; r2 < 4; ++r2) {
        Pack2 a, c;
        a.e[0] = (_Float16)p0[2 * r2]; a.e[1] = (_Float16)p0[2 * r2 + 1];
        c.e[0] = (_Float16)p1[2 * r2]; c.e[1] = (_Float16)p1[2 * r2 + 1];
        ldsPu[wv][nt][rl][4 * hi + r2]     = a.u;
        ldsPu[wv][nt][rl][8 + 4 * hi + r2] = c.u;
      }
      if (hi == 0) ldsC[wv][nt][rl] = corr;
    }

    // V B-fragments (shared by both n-tiles).
    v16h bV[4];
#pragma unroll
    for (int t = 0; t < 4; ++t)
      bV[t] = load_frag_b(v_u, t * 16, N_ / 2, mt, lane);

#pragma unroll
    for (int nt = 0; nt < 2; ++nt) {
      // Rescale O by corr of its row n = r + 8*hi (broadcast via LDS).
      float cf[8];
#pragma unroll
      for (int r = 0; r < 8; ++r) cf[r] = ldsC[wv][nt][r + 8 * hi];
#pragma unroll
      for (int t = 0; t < 4; ++t)
#pragma unroll
        for (int r = 0; r < 8; ++r) acc[nt][t][r] *= cf[r];

      // Re-read P as 16x32 A fragment (row n = rl, K = m).
      FragU fp;
#pragma unroll
      for (int v = 0; v < 8; ++v) {
        int cu = ((v & 4) ? 8 : 0) + (hi ? 4 : 0) + (v & 3);
        fp.u[v] = ldsPu[wv][nt][rl][cu];
      }
#pragma unroll
      for (int t = 0; t < 4; ++t)
        acc[nt][t] = wmma16(fp.h, bV[t], acc[nt][t]);
    }
  }

  // Normalize by l (broadcast through LDS) and store o_h [row][64] f16.
  if (hi == 0) { ldsC[wv][0][rl] = lrun[0]; ldsC[wv][1][rl] = lrun[1]; }
#pragma unroll
  for (int nt = 0; nt < 2; ++nt) {
#pragma unroll
    for (int r = 0; r < 8; ++r) {
      float inv = 1.0f / ldsC[wv][nt][r + 8 * hi];
      int grow = row0 + nt * 16 + r + 8 * hi;
#pragma unroll
      for (int t = 0; t < 4; ++t)
        oh[(long)grow * C_ + t * 16 + rl] = (_Float16)(acc[nt][t][r] * inv);
    }
  }
}

// ---------------------------------------------------------------------------
// Host-side launcher
// ---------------------------------------------------------------------------
extern "C" void kernel_launch(void* const* d_in, const int* in_sizes, int n_in,
                              void* d_out, int out_size, void* d_ws, size_t ws_size,
                              hipStream_t stream) {
  (void)in_sizes; (void)n_in; (void)out_size; (void)ws_size;

  const float* x         = (const float*)d_in[0];
  const float* cta_ln_g  = (const float*)d_in[1];
  const float* cta_ln_b  = (const float*)d_in[2];
  const float* q_w       = (const float*)d_in[3];
  const float* q_b       = (const float*)d_in[4];
  const float* k_w       = (const float*)d_in[5];
  const float* k_b       = (const float*)d_in[6];
  const float* v_w       = (const float*)d_in[7];
  const float* v_b       = (const float*)d_in[8];
  const float* cta_out_w = (const float*)d_in[9];
  const float* cta_out_b = (const float*)d_in[10];
  const float* nle_ln_g  = (const float*)d_in[11];
  const float* nle_ln_b  = (const float*)d_in[12];
  const float* b1_w1     = (const float*)d_in[13];
  const float* b1_b1     = (const float*)d_in[14];
  const float* b1_w2     = (const float*)d_in[15];
  const float* b1_b2     = (const float*)d_in[16];
  const float* b2_w1     = (const float*)d_in[17];
  const float* b2_b1     = (const float*)d_in[18];
  const float* b2_w2     = (const float*)d_in[19];
  const float* b2_b2     = (const float*)d_in[20];
  const float* nle_out_w = (const float*)d_in[21];
  const float* nle_out_b = (const float*)d_in[22];

  char* wsb = (char*)d_ws;
  size_t off = 0;
  auto alloc = [&](size_t bytes) -> void* {
    void* p = wsb + off;
    off += (bytes + 255) & ~(size_t)255;
    return p;
  };

  float*    xn    = (float*)   alloc((size_t)ROWS * C_ * 4);   // [b][c][n]
  _Float16* xnh   = (_Float16*)alloc((size_t)ROWS * C_ * 2);   // [row][c]
  _Float16* qh    = (_Float16*)alloc((size_t)ROWS * C_ * 2);   // [row][c], pre-scaled 1/8
  _Float16* khb   = (_Float16*)alloc((size_t)ROWS * C_ * 2);   // [b][m][c]
  _Float16* vhb   = (_Float16*)alloc((size_t)ROWS * C_ * 2);   // [b][c][m]
  _Float16* ohb   = (_Float16*)alloc((size_t)ROWS * C_ * 2);   // [row][c]
  float*    xatt  = (float*)   alloc((size_t)ROWS * C_ * 4);   // [row][c]
  _Float16* xn2h  = (_Float16*)alloc((size_t)ROWS * C_ * 2);   // [row][c]
  _Float16* h1h   = (_Float16*)alloc((size_t)ROWS * 128 * 2);  // [row][128]
  _Float16* h2h   = (_Float16*)alloc((size_t)ROWS * 128 * 2);
  _Float16* g1h   = (_Float16*)alloc((size_t)ROWS * 128 * 2);
  _Float16* prodh = (_Float16*)alloc((size_t)ROWS * 128 * 2);
  _Float16* qwh   = (_Float16*)alloc(64 * 64 * 2);
  _Float16* ctawh = (_Float16*)alloc(64 * 64 * 2);
  _Float16* b1wh  = (_Float16*)alloc(128 * 64 * 2);
  _Float16* b2wh  = (_Float16*)alloc(128 * 64 * 2);
  _Float16* nlewh = (_Float16*)alloc(64 * 128 * 2);

  // 0) weights -> fp16 ([co][ci] row-major); q_w folded with 1/sqrt(C)=0.125
  f32_to_f16_kernel<<<16, 256, 0, stream>>>(q_w,       qwh,   64 * 64,  0.125f);
  f32_to_f16_kernel<<<16, 256, 0, stream>>>(cta_out_w, ctawh, 64 * 64,  1.0f);
  f32_to_f16_kernel<<<32, 256, 0, stream>>>(b1_w1,     b1wh,  128 * 64, 1.0f);
  f32_to_f16_kernel<<<32, 256, 0, stream>>>(b2_w1,     b2wh,  128 * 64, 1.0f);
  f32_to_f16_kernel<<<32, 256, 0, stream>>>(nle_out_w, nlewh, 64 * 128, 1.0f);

  // 1) LayerNorm 1
  ln1_kernel<<<ROWS / 256, 256, 0, stream>>>(x, cta_ln_g, cta_ln_b, xn, xnh);

  // 2) q projection (WMMA, bias also scaled by 0.125), k/v depthwise 3x3
  gemm_wmma_kernel<64, 64, 0, 0><<<ROWS / 64, 128, 0, stream>>>(xnh, qwh, q_b, qh, nullptr, 0.125f);
  dwconv64_kernel<<<(B_ * C_ * N_) / 256, 256, 0, stream>>>(xn, k_w, k_b, khb, 1);
  dwconv64_kernel<<<(B_ * C_ * N_) / 256, 256, 0, stream>>>(xn, v_w, v_b, vhb, 0);

  // 3) flash attention (WMMA, register-axis online softmax, 32 rows/wave)
  flash_attn_kernel<<<ROWS / 128, 128, 0, stream>>>(qh, khb, vhb, ohb);

  // 4) cta_out 1x1 (WMMA) + residual x  -> x_att f32 [row][c]
  gemm_wmma_kernel<64, 64, 1, 1><<<ROWS / 64, 128, 0, stream>>>(ohb, ctawh, cta_out_b, xatt, x, 1.0f);

  // 5) LayerNorm 2
  ln2_kernel<<<ROWS / 256, 256, 0, stream>>>(xatt, nle_ln_g, nle_ln_b, xn2h);

  // 6) two MLP branches: 1x1 (WMMA) -> dw3x3 + GELU (branch 2 fuses the multiply)
  gemm_wmma_kernel<64, 128, 0, 0><<<ROWS / 64, 128, 0, stream>>>(xn2h, b1wh, b1_b1, h1h, nullptr, 1.0f);
  gemm_wmma_kernel<64, 128, 0, 0><<<ROWS / 64, 128, 0, stream>>>(xn2h, b2wh, b2_b1, h2h, nullptr, 1.0f);
  dwconv128_gelu_kernel<<<(ROWS * 128) / 256, 256, 0, stream>>>(h1h, b1_w2, b1_b2, nullptr, g1h);
  dwconv128_gelu_kernel<<<(ROWS * 128) / 256, 256, 0, stream>>>(h2h, b2_w2, b2_b2, g1h, prodh);

  // 7) nle_out 1x1 (WMMA, K=128) + residual x_att, stored as [b][c][h][w]
  gemm_wmma_kernel<128, 64, 2, 2><<<ROWS / 64, 128, 0, stream>>>(prodh, nlewh, nle_out_b,
                                                                 (float*)d_out, xatt, 1.0f);
}
